// GraphAttention_29927332118760
// MI455X (gfx1250) — compile-verified, matrix operand order checked by reading, not run on hardware
//
#include <hip/hip_runtime.h>
#include <math.h>

typedef __attribute__((ext_vector_type(16))) _Float16 v16h;
typedef __attribute__((ext_vector_type(8)))  float    v8f;
typedef __attribute__((ext_vector_type(4)))  float    v4f;

#define N_DST 16384
#define N_SRC 16384
#define NEDGE 262144
#define NW    2          // waves per block in edge/node/final kernels

// packed-weight (f16) element offsets inside workspace
static constexpr int OFF_RW1 = 0;      // rad_w1 64x64   -> 4096
static constexpr int OFF_RW2 = 4096;   // rad_w2 64x224  -> 14336
static constexpr int OFF_L0  = 18432;  // lin0   96x128  -> 12288
static constexpr int OFF_L1  = 30720;  // lin1   128x32  -> 4096
static constexpr int OFF_WD0 = 34816;  // W_dst0 64x64   -> 4096
static constexpr int OFF_WD1 = 38912;  // W_dst1 32x32   -> 1024
static constexpr int OFF_WS0 = 39936;  // W_src0 64x64   -> 4096
static constexpr int OFF_WS1 = 44032;  // W_src1 32x32   -> 1024
static constexpr int OFF_P0  = 45056;  // proj0  64x64   -> 4096
static constexpr int OFF_P1  = 49152;  // proj1  32x32   -> 1024
static constexpr size_t PACK_BYTES = 131072; // 128KB region, >= 50176*2

__device__ __forceinline__ int kmap(int lane, int j) {
    // A/B fragment K index for element j of lane: runs of 8, split by lane-half
    return ((j >> 3) << 4) | ((lane >> 4) << 3) | (j & 7);
}

__device__ __forceinline__ v8f wmma16(v16h a, v16h b, v8f c) {
    return __builtin_amdgcn_wmma_f32_16x16x32_f16(false, a, false, b, (short)0, c, false, false);
}

__device__ __forceinline__ v16h loadB(const _Float16* pack, int frag, int lane) {
    return *(const v16h*)(pack + ((size_t)(frag * 32 + lane) << 4));
}

__device__ __forceinline__ v8f splat8(float x) {
    v8f r; for (int i = 0; i < 8; ++i) r[i] = x; return r;
}

__device__ __forceinline__ void atomicMaxF(float* addr, float val) {
    int* ia = (int*)addr;
    int cur = *ia;
    while (__int_as_float(cur) < val) {
        int assumed = cur;
        cur = atomicCAS(ia, assumed, __float_as_int(val));
        if (cur == assumed) break;
    }
}

// ---------------- weight packing into B-fragment order ----------------
__global__ void pack_kernel(const float* __restrict__ src, _Float16* __restrict__ dst,
                            int K, int N) {
    int tid = blockIdx.x * blockDim.x + threadIdx.x;
    if (tid >= K * N) return;
    int j = tid & 15;
    int L = (tid >> 4) & 31;
    int r = tid >> 9;          // kb*NT + nt
    int NT = N >> 4;
    int nt = r % NT, kb = r / NT;
    int Ki = kb * 32 + kmap(L, j);
    int Ni = nt * 16 + (L & 15);
    dst[tid] = (_Float16)src[(size_t)Ki * N + Ni];
}

// ---------------- init segment state + accumulators ----------------
__global__ void init_kernel(float* segmax, float* segden, float* aggS, float* aggV) {
    int tid = blockIdx.x * blockDim.x + threadIdx.x;
    if (tid < N_DST * 4) { segmax[tid] = -3.0e38f; segden[tid] = 0.f; }
    if (tid < N_DST * 64) aggS[tid] = 0.f;
    if (tid < N_DST * 96) aggV[tid] = 0.f;
}

// ---------------- per-node input transforms ----------------
__global__ void node_kernel(const float* __restrict__ dst_input,
                            const float* __restrict__ src_input,
                            const _Float16* __restrict__ packW,
                            float* md_s, float* md_v, float* msr_s, float* msr_v) {
    int wid  = (blockIdx.x * blockDim.x + threadIdx.x) >> 5;
    int lane = threadIdx.x & 31;
    int lr = lane & 15, hi = lane >> 4;
    int half  = wid >= (N_DST / 16);
    int nbase = (half ? wid - N_DST / 16 : wid) * 16;
    const float* in  = half ? src_input : dst_input;
    const _Float16* pw0 = packW + (half ? OFF_WS0 : OFF_WD0);
    const _Float16* pw1 = packW + (half ? OFF_WS1 : OFF_WD1);
    float* outS = half ? msr_s : md_s;
    float* outV = half ? msr_v : md_v;
    const float* row = in + (size_t)(nbase + lr) * 160;

    v16h a0, a1;
    for (int j = 0; j < 16; ++j) {
        a0[j] = (_Float16)row[kmap(lane, j)];
        a1[j] = (_Float16)row[32 + kmap(lane, j)];
    }
    for (int nt = 0; nt < 4; ++nt) {
        v8f acc = {};
        acc = wmma16(a0, loadB(pw0, 0 * 4 + nt, lane), acc);
        acc = wmma16(a1, loadB(pw0, 1 * 4 + nt, lane), acc);
        for (int v = 0; v < 8; ++v)
            outS[(size_t)(nbase + v + hi * 8) * 64 + nt * 16 + lr] = acc[v];
    }
    for (int c = 0; c < 3; ++c) {
        v16h av;
        for (int j = 0; j < 16; ++j)
            av[j] = (_Float16)row[64 + kmap(lane, j) * 3 + c];
        for (int nt = 0; nt < 2; ++nt) {
            v8f acc = {};
            acc = wmma16(av, loadB(pw1, nt, lane), acc);
            for (int v = 0; v < 8; ++v)
                outV[(size_t)(nbase + v + hi * 8) * 96 + (nt * 16 + lr) * 3 + c] = acc[v];
        }
    }
}

// ---------------- per-edge pipeline (PHASE 0: logits, PHASE 1: aggregate) ----------------
template <int PHASE>
__global__ void edge_kernel(const float* __restrict__ scalars,
                            const float* __restrict__ src_attr,
                            const int* __restrict__ dst_index,
                            const int* __restrict__ src_index,
                            const float* __restrict__ rad_b1,
                            const float* __restrict__ rad_b2,
                            const float* __restrict__ lin0_b,
                            const float* __restrict__ alpha_dot,
                            const _Float16* __restrict__ packW,
                            const float* __restrict__ md_s, const float* __restrict__ md_v,
                            const float* __restrict__ msr_s, const float* __restrict__ msr_v,
                            float* logits, float* segmax, const float* segden,
                            float* aggS, float* aggV) {
    __shared__ _Float16 sh_h[NW][16][64];
    __shared__ float    sh_w[NW][16][224];
    __shared__ _Float16 sh_dtps[NW][16][96];
    __shared__ _Float16 sh_dtpv[NW][3][16][128];

    int w    = threadIdx.x >> 5;
    int lane = threadIdx.x & 31;
    int lr = lane & 15, hi = lane >> 4;
    int ebase = (blockIdx.x * NW + w) * 16;

    // --- radial layer 1: scalars(16x64) @ rad_w1(64x64), silu ---
    const float* srow = scalars + (size_t)(ebase + lr) * 64;
    v16h ra0, ra1;
    for (int j = 0; j < 16; ++j) {
        ra0[j] = (_Float16)srow[kmap(lane, j)];
        ra1[j] = (_Float16)srow[32 + kmap(lane, j)];
    }
    const _Float16* prw1 = packW + OFF_RW1;
    for (int nt = 0; nt < 4; ++nt) {
        v8f acc = splat8(rad_b1[nt * 16 + lr]);
        acc = wmma16(ra0, loadB(prw1, 0 * 4 + nt, lane), acc);
        acc = wmma16(ra1, loadB(prw1, 1 * 4 + nt, lane), acc);
        for (int v = 0; v < 8; ++v) {
            float x = acc[v];
            sh_h[w][v + hi * 8][nt * 16 + lr] = (_Float16)(x / (1.f + __expf(-x)));
        }
    }
    __syncthreads();

    // --- radial layer 2: h(16x64) @ rad_w2(64x224) ---
    v16h h0, h1;
    {
        const _Float16* hr = sh_h[w][lr];
        for (int j = 0; j < 8; ++j) {
            h0[j] = hr[hi * 8 + j];      h0[8 + j] = hr[16 + hi * 8 + j];
            h1[j] = hr[32 + hi * 8 + j]; h1[8 + j] = hr[48 + hi * 8 + j];
        }
    }
    const _Float16* prw2 = packW + OFF_RW2;
    for (int nt = 0; nt < 14; ++nt) {
        if (PHASE == 0 && !(nt < 4 || nt == 10 || nt == 11)) continue; // only w00,w110 needed
        v8f acc = splat8(rad_b2[nt * 16 + lr]);
        acc = wmma16(h0, loadB(prw2, 0 * 14 + nt, lane), acc);
        acc = wmma16(h1, loadB(prw2, 1 * 14 + nt, lane), acc);
        for (int v = 0; v < 8; ++v)
            sh_w[w][v + hi * 8][nt * 16 + lr] = acc[v];
    }
    __syncthreads();

    // --- tensor product: lane handles edge lr, lane-half splits channels ---
    {
        int e = ebase + lr;
        int d = dst_index[e], s = src_index[e];
        v4f attr = *(const v4f*)(src_attr + (size_t)e * 4);
        float a0v = attr[0], avx = attr[1], avy = attr[2], avz = attr[3];
        const float* wrow = sh_w[w][lr];
        const float* pds = md_s  + (size_t)d * 64;
        const float* pss = msr_s + (size_t)s * 64;
        const float* pdv = md_v  + (size_t)d * 96;
        const float* psv = msr_v + (size_t)s * 96;
        for (int c0 = 0; c0 < 32; ++c0) {
            int c = hi * 32 + c0;
            float ms = pds[c] + pss[c];
            sh_dtps[w][lr][c] = (_Float16)(ms * a0v * wrow[c]);           // out00
            if (PHASE == 1) {
                float t = ms * wrow[64 + c];                               // out01
                sh_dtpv[w][0][lr][c] = (_Float16)(t * avx);
                sh_dtpv[w][1][lr][c] = (_Float16)(t * avy);
                sh_dtpv[w][2][lr][c] = (_Float16)(t * avz);
            }
        }
        for (int u0 = 0; u0 < 16; ++u0) {
            int u = hi * 16 + u0;
            float mx = pdv[u * 3 + 0] + psv[u * 3 + 0];
            float my = pdv[u * 3 + 1] + psv[u * 3 + 1];
            float mz = pdv[u * 3 + 2] + psv[u * 3 + 2];
            float dotv = mx * avx + my * avy + mz * avz;
            sh_dtps[w][lr][64 + u] = (_Float16)(dotv * wrow[160 + u] * 0.5773502691896258f); // out110
            if (PHASE == 1) {
                float w10 = wrow[128 + u] * a0v;                           // out10
                sh_dtpv[w][0][lr][64 + u] = (_Float16)(mx * w10);
                sh_dtpv[w][1][lr][64 + u] = (_Float16)(my * w10);
                sh_dtpv[w][2][lr][64 + u] = (_Float16)(mz * w10);
                float w111 = wrow[192 + u] * 0.7071067811865476f;          // out111
                sh_dtpv[w][0][lr][96 + u] = (_Float16)((my * avz - mz * avy) * w111);
                sh_dtpv[w][1][lr][96 + u] = (_Float16)((mz * avx - mx * avz) * w111);
                sh_dtpv[w][2][lr][96 + u] = (_Float16)((mx * avy - my * avx) * w111);
            }
        }
    }
    __syncthreads();

    // --- lin0: dtp_s(16x96) @ lin0(96x128) ---
    v16h ds0, ds1, ds2;
    {
        const _Float16* dr = sh_dtps[w][lr];
        for (int j = 0; j < 8; ++j) {
            ds0[j] = dr[hi * 8 + j];      ds0[8 + j] = dr[16 + hi * 8 + j];
            ds1[j] = dr[32 + hi * 8 + j]; ds1[8 + j] = dr[48 + hi * 8 + j];
            ds2[j] = dr[64 + hi * 8 + j]; ds2[8 + j] = dr[80 + hi * 8 + j];
        }
    }
    const _Float16* pl0 = packW + OFF_L0;
    for (int nt = (PHASE == 0 ? 0 : 1); nt < 8; nt += 2) {
        v8f acc = splat8(lin0_b[nt * 16 + lr]);
        acc = wmma16(ds0, loadB(pl0, 0 * 8 + nt, lane), acc);
        acc = wmma16(ds1, loadB(pl0, 1 * 8 + nt, lane), acc);
        acc = wmma16(ds2, loadB(pl0, 2 * 8 + nt, lane), acc);
        if (PHASE == 0) {
            int h = nt >> 1;
            float ad = alpha_dot[h * 16 + lr];
            for (int v = 0; v < 8; ++v) {
                float x = acc[v];
                float g = 0.6f * x + 0.4f * x * (2.f / (1.f + __expf(-x)) - 1.f);
                float t = g * ad;
                for (int m = 1; m < 16; m <<= 1) t += __shfl_xor(t, m, 32);
                if (lr == 0) {
                    int e = ebase + v + hi * 8;
                    logits[(size_t)e * 4 + h] = t;
                    int d = dst_index[e];
                    atomicMaxF(&segmax[d * 4 + h], t);
                }
            }
        } else {
            int h = (nt - 1) >> 1;
            for (int v = 0; v < 8; ++v) {
                int e = ebase + v + hi * 8;
                int d = dst_index[e];
                float attn = __expf(logits[(size_t)e * 4 + h] - segmax[d * 4 + h]) / segden[d * 4 + h];
                atomicAdd(&aggS[(size_t)d * 64 + h * 16 + lr], acc[v] * attn);
            }
        }
    }

    if (PHASE == 1) {
        // --- lin1: dtp_v(16x128) @ lin1(128x32), per component ---
        const _Float16* pl1 = packW + OFF_L1;
        for (int c = 0; c < 3; ++c) {
            v16h dv0, dv1, dv2, dv3;
            const _Float16* dr = sh_dtpv[w][c][lr];
            for (int j = 0; j < 8; ++j) {
                dv0[j] = dr[hi * 8 + j];       dv0[8 + j] = dr[16 + hi * 8 + j];
                dv1[j] = dr[32 + hi * 8 + j];  dv1[8 + j] = dr[48 + hi * 8 + j];
                dv2[j] = dr[64 + hi * 8 + j];  dv2[8 + j] = dr[80 + hi * 8 + j];
                dv3[j] = dr[96 + hi * 8 + j];  dv3[8 + j] = dr[112 + hi * 8 + j];
            }
            for (int nt = 0; nt < 2; ++nt) {
                v8f acc = {};
                acc = wmma16(dv0, loadB(pl1, 0 * 2 + nt, lane), acc);
                acc = wmma16(dv1, loadB(pl1, 1 * 2 + nt, lane), acc);
                acc = wmma16(dv2, loadB(pl1, 2 * 2 + nt, lane), acc);
                acc = wmma16(dv3, loadB(pl1, 3 * 2 + nt, lane), acc);
                int u2 = nt * 16 + lr;
                int h  = u2 >> 3;
                for (int v = 0; v < 8; ++v) {
                    int e = ebase + v + hi * 8;
                    int d = dst_index[e];
                    float attn = __expf(logits[(size_t)e * 4 + h] - segmax[d * 4 + h]) / segden[d * 4 + h];
                    atomicAdd(&aggV[(size_t)d * 96 + u2 * 3 + c], acc[v] * attn);
                }
            }
        }
    }
}

// ---------------- softmax denominator ----------------
__global__ void den_kernel(const float* __restrict__ logits, const int* __restrict__ dst_index,
                           const float* __restrict__ segmax, float* segden) {
    int tid = blockIdx.x * blockDim.x + threadIdx.x;
    if (tid >= NEDGE * 4) return;
    int e = tid >> 2, h = tid & 3;
    int d = dst_index[e];
    atomicAdd(&segden[d * 4 + h], __expf(logits[tid] - segmax[d * 4 + h]));
}

// ---------------- output projections ----------------
__global__ void final_kernel(const float* __restrict__ aggS, const float* __restrict__ aggV,
                             const _Float16* __restrict__ packW,
                             const float* __restrict__ proj0_b, float* __restrict__ out) {
    int wid  = (blockIdx.x * blockDim.x + threadIdx.x) >> 5;
    int lane = threadIdx.x & 31;
    int lr = lane & 15, hi = lane >> 4;
    int nbase = wid * 16;

    const float* row = aggS + (size_t)(nbase + lr) * 64;
    v16h a0, a1;
    for (int j = 0; j < 16; ++j) {
        a0[j] = (_Float16)row[kmap(lane, j)];
        a1[j] = (_Float16)row[32 + kmap(lane, j)];
    }
    const _Float16* pp0 = packW + OFF_P0;
    for (int nt = 0; nt < 4; ++nt) {
        v8f acc = splat8(proj0_b[nt * 16 + lr]);
        acc = wmma16(a0, loadB(pp0, 0 * 4 + nt, lane), acc);
        acc = wmma16(a1, loadB(pp0, 1 * 4 + nt, lane), acc);
        for (int v = 0; v < 8; ++v)
            out[(size_t)(nbase + v + hi * 8) * 160 + nt * 16 + lr] = acc[v];
    }
    const _Float16* pp1 = packW + OFF_P1;
    const float* vrow = aggV + (size_t)(nbase + lr) * 96;
    for (int c = 0; c < 3; ++c) {
        v16h av;
        for (int j = 0; j < 16; ++j)
            av[j] = (_Float16)vrow[kmap(lane, j) * 3 + c];
        for (int nt = 0; nt < 2; ++nt) {
            v8f acc = {};
            acc = wmma16(av, loadB(pp1, nt, lane), acc);
            for (int v = 0; v < 8; ++v)
                out[(size_t)(nbase + v + hi * 8) * 160 + 64 + (nt * 16 + lr) * 3 + c] = acc[v];
        }
    }
}

extern "C" void kernel_launch(void* const* d_in, const int* in_sizes, int n_in,
                              void* d_out, int out_size, void* d_ws, size_t ws_size,
                              hipStream_t stream) {
    (void)in_sizes; (void)n_in; (void)out_size; (void)ws_size;
    const float* dst_input = (const float*)d_in[0];
    const float* src_input = (const float*)d_in[1];
    const float* src_attr  = (const float*)d_in[2];
    const float* scalars   = (const float*)d_in[3];
    const int*   dst_index = (const int*)d_in[4];
    const int*   src_index = (const int*)d_in[5];
    const float* W_dst0 = (const float*)d_in[6];
    const float* W_dst1 = (const float*)d_in[7];
    const float* W_src0 = (const float*)d_in[8];
    const float* W_src1 = (const float*)d_in[9];
    const float* rad_w1 = (const float*)d_in[10];
    const float* rad_b1 = (const float*)d_in[11];
    const float* rad_w2 = (const float*)d_in[12];
    const float* rad_b2 = (const float*)d_in[13];
    const float* lin0   = (const float*)d_in[14];
    const float* lin0_b = (const float*)d_in[15];
    const float* lin1   = (const float*)d_in[16];
    const float* alpha_dot = (const float*)d_in[17];
    const float* proj0   = (const float*)d_in[18];
    const float* proj0_b = (const float*)d_in[19];
    const float* proj1   = (const float*)d_in[20];
    float* out = (float*)d_out;

    _Float16* packW = (_Float16*)d_ws;
    float* f = (float*)((char*)d_ws + PACK_BYTES);
    float* md_s  = f; f += (size_t)N_DST * 64;
    float* md_v  = f; f += (size_t)N_DST * 96;
    float* msr_s = f; f += (size_t)N_DST * 64;
    float* msr_v = f; f += (size_t)N_DST * 96;
    float* logits = f; f += (size_t)NEDGE * 4;
    float* segmax = f; f += (size_t)N_DST * 4;
    float* segden = f; f += (size_t)N_DST * 4;
    float* aggS  = f; f += (size_t)N_DST * 64;
    float* aggV  = f; f += (size_t)N_DST * 96;

    auto packL = [&](const float* src, int off, int K, int N) {
        int total = K * N;
        pack_kernel<<<(total + 255) / 256, 256, 0, stream>>>(src, packW + off, K, N);
    };
    packL(rad_w1, OFF_RW1, 64, 64);
    packL(rad_w2, OFF_RW2, 64, 224);
    packL(lin0,   OFF_L0,  96, 128);
    packL(lin1,   OFF_L1, 128, 32);
    packL(W_dst0, OFF_WD0, 64, 64);
    packL(W_dst1, OFF_WD1, 32, 32);
    packL(W_src0, OFF_WS0, 64, 64);
    packL(W_src1, OFF_WS1, 32, 32);
    packL(proj0,  OFF_P0,  64, 64);
    packL(proj1,  OFF_P1,  32, 32);

    int initN = N_DST * 96;
    init_kernel<<<(initN + 255) / 256, 256, 0, stream>>>(segmax, segden, aggS, aggV);

    node_kernel<<<(2 * N_DST / 16) * 32 / (NW * 32), NW * 32, 0, stream>>>(
        dst_input, src_input, packW, md_s, md_v, msr_s, msr_v);

    int edgeBlocks = NEDGE / 16 / NW;
    edge_kernel<0><<<edgeBlocks, NW * 32, 0, stream>>>(
        scalars, src_attr, dst_index, src_index, rad_b1, rad_b2, lin0_b, alpha_dot,
        packW, md_s, md_v, msr_s, msr_v, logits, segmax, segden, aggS, aggV);

    den_kernel<<<(NEDGE * 4 + 255) / 256, 256, 0, stream>>>(logits, dst_index, segmax, segden);

    edge_kernel<1><<<edgeBlocks, NW * 32, 0, stream>>>(
        scalars, src_attr, dst_index, src_index, rad_b1, rad_b2, lin0_b, alpha_dot,
        packW, md_s, md_v, msr_s, msr_v, logits, segmax, segden, aggS, aggV);

    final_kernel<<<(N_DST / 16) / NW, NW * 32, 0, stream>>>(aggS, aggV, packW, proj0_b, out);
}
